// GRUCellSeq_82222853914664
// MI455X (gfx1250) — matrix operand order repaired and checked by reading, not compiled, not run on hardware
//
#include <hip/hip_runtime.h>

#define T_SEQ 512
#define BATCH 128
#define DIN   512
#define DHID  512
#define G3    (3 * DHID)   // 1536

typedef __bf16 bf16_t;
typedef __bf16 v16bf __attribute__((ext_vector_type(16)));
typedef __bf16 v8bf  __attribute__((ext_vector_type(8)));
typedef float  v8f   __attribute__((ext_vector_type(8)));

// ---------------------------------------------------------------------------
// fp32 -> bf16 conversion (vectorized x4)
// ---------------------------------------------------------------------------
__global__ void cvt_f32_to_bf16(const float* __restrict__ src,
                                bf16_t* __restrict__ dst, int n4) {
    int i = blockIdx.x * blockDim.x + threadIdx.x;
    if (i < n4) {
        const float4 v = reinterpret_cast<const float4*>(src)[i];
        dst[4 * i + 0] = (bf16_t)v.x;
        dst[4 * i + 1] = (bf16_t)v.y;
        dst[4 * i + 2] = (bf16_t)v.z;
        dst[4 * i + 3] = (bf16_t)v.w;
    }
}

__global__ void zero_bf16(bf16_t* __restrict__ dst, int n) {
    int i = blockIdx.x * blockDim.x + threadIdx.x;
    if (i < n) dst[i] = (bf16_t)0.0f;
}

// ---------------------------------------------------------------------------
// A-tile loader: 16x32 bf16 tile of a row-major matrix.
// Lane l holds row (l&15); element i -> K0+(l>>4)*16+i (matches B). One 32B load.
// ---------------------------------------------------------------------------
__device__ __forceinline__ v16bf load_a_tile(const bf16_t* __restrict__ rowp,
                                             int k0, int half) {
    return *reinterpret_cast<const v16bf*>(rowp + k0 + half * 16);
}

// B-tile loader: 32x16 bf16 tile where B[k][n] = W[n][k] (W row-major).
// Lane l holds column n = (l&15); K = k0 + (l>>4)*16 + i. One 32B load.
__device__ __forceinline__ v16bf load_b_tile(const bf16_t* __restrict__ wrow,
                                             int k0, int half) {
    return *reinterpret_cast<const v16bf*>(wrow + k0 + half * 16);
}

__device__ __forceinline__ float sigmoid_f(float x) {
    return 1.0f / (1.0f + __expf(-x));
}

// ---------------------------------------------------------------------------
// gi = x_bf16 (T*B,512) @ W_ih_bf16^T (512,1536) + b_ih   -> fp32 (T*B,1536)
// Register-blocked: one wave owns a 64x64 output block (4x4 WMMA tiles),
// so each A/B fragment is reused 4x -> 16 v_wmma per 8 vector loads.
// ---------------------------------------------------------------------------
__global__ __launch_bounds__(256) void gi_gemm_kernel(
        const bf16_t* __restrict__ xb,    // (T*B, DIN)
        const bf16_t* __restrict__ wih,   // (G3, DIN) row-major
        const float* __restrict__ bih,    // (G3)
        float* __restrict__ gi)           // (T*B, G3)
{
    const int lane  = threadIdx.x & 31;
    const int wave  = threadIdx.x >> 5;
    const int wtile = blockIdx.x * 8 + wave;   // 64x64 block index
    const int nBlkN = G3 / 64;                 // 24
    const int mt = wtile / nBlkN;              // 0..1023
    const int nt = wtile - mt * nBlkN;
    const int row0 = mt * 64, col0 = nt * 64;
    const int half = lane >> 4, l16 = lane & 15;

    const bf16_t* arow[4];
    const bf16_t* brow[4];
#pragma unroll
    for (int i = 0; i < 4; ++i) {
        arow[i] = xb  + (size_t)(row0 + i * 16 + l16) * DIN;
        brow[i] = wih + (size_t)(col0 + i * 16 + l16) * DIN;
    }

    v8f acc[4][4] = {};
#pragma unroll 2
    for (int k0 = 0; k0 < DIN; k0 += 32) {
        v16bf a[4], b[4];
#pragma unroll
        for (int i = 0; i < 4; ++i) {
            a[i] = load_a_tile(arow[i], k0, half);
            b[i] = load_b_tile(brow[i], k0, half);
        }
#pragma unroll
        for (int mi = 0; mi < 4; ++mi)
#pragma unroll
            for (int ni = 0; ni < 4; ++ni)
                acc[mi][ni] = __builtin_amdgcn_wmma_f32_16x16x32_bf16(
                    false, a[mi], false, b[ni], (short)0, acc[mi][ni],
                    false, false);
    }

    // Epilogue: C/D layout -> VGPR r holds row 8*half + r, column l16.
#pragma unroll
    for (int ni = 0; ni < 4; ++ni) {
        const int j = col0 + ni * 16 + l16;
        const float bias = bih[j];
#pragma unroll
        for (int mi = 0; mi < 4; ++mi) {
            float* orow = gi + (size_t)(row0 + mi * 16 + half * 8) * G3 + j;
#pragma unroll
            for (int r = 0; r < 8; ++r)
                orow[(size_t)r * G3] = acc[mi][ni][r] + bias;
        }
    }
}

// ---------------------------------------------------------------------------
// One GRU timestep.  gh = h_prev @ W_hh^T (+ b_hh folded into epilogue),
// then fused gates.  8x32 = 256 tiles of 16x16 -> 32 blocks x 8 waves.
// Kept one-tile-per-wave: tiny serial-critical-path GEMM, parallelism and
// latency matter more than reuse; one A-tile already feeds 3 WMMAs (r,z,n).
// ---------------------------------------------------------------------------
__global__ __launch_bounds__(256) void gru_step_kernel(
        const bf16_t* __restrict__ hprev_b,  // (B, DHID) bf16 (zeros at t=0)
        const float* __restrict__ hprev_f,   // (B, DHID) fp32, null at t=0
        const bf16_t* __restrict__ whh,      // (G3, DHID) row-major bf16
        const float* __restrict__ bhh,       // (G3)
        const float* __restrict__ gi_t,      // (B, G3) fp32
        float* __restrict__ out_t,           // (B, DHID) fp32
        bf16_t* __restrict__ hnext_b,        // (B, DHID) bf16
        float* __restrict__ hlast)           // null unless t == T-1
{
    const int lane = threadIdx.x & 31;
    const int wave = threadIdx.x >> 5;
    const int tile = blockIdx.x * 8 + wave;   // 0..255
    const int nt = tile & 31;                 // 32 column tiles
    const int mt = tile >> 5;                 // 8 row tiles
    const int row0 = mt * 16, col0 = nt * 16;
    const int half = lane >> 4, l16 = lane & 15;

    const bf16_t* arow = hprev_b + (size_t)(row0 + l16) * DHID;
    const bf16_t* wr = whh + (size_t)(col0 + l16) * DHID;             // r gate
    const bf16_t* wz = wr + (size_t)DHID * DHID;                      // z gate
    const bf16_t* wn = wz + (size_t)DHID * DHID;                      // n gate

    v8f accr = {}, accz = {}, accn = {};
#pragma unroll 2
    for (int k0 = 0; k0 < DHID; k0 += 32) {
        v16bf a  = load_a_tile(arow, k0, half);
        v16bf br = load_b_tile(wr, k0, half);
        v16bf bz = load_b_tile(wz, k0, half);
        v16bf bn = load_b_tile(wn, k0, half);
        accr = __builtin_amdgcn_wmma_f32_16x16x32_bf16(
                   false, a, false, br, (short)0, accr, false, false);
        accz = __builtin_amdgcn_wmma_f32_16x16x32_bf16(
                   false, a, false, bz, (short)0, accz, false, false);
        accn = __builtin_amdgcn_wmma_f32_16x16x32_bf16(
                   false, a, false, bn, (short)0, accn, false, false);
    }

    const int j = col0 + l16;
    const float br_b = bhh[j];
    const float bz_b = bhh[DHID + j];
    const float bn_b = bhh[2 * DHID + j];

#pragma unroll
    for (int r = 0; r < 8; ++r) {
        const int b = row0 + half * 8 + r;
        const float gir = gi_t[(size_t)b * G3 + j];
        const float giz = gi_t[(size_t)b * G3 + DHID + j];
        const float gin = gi_t[(size_t)b * G3 + 2 * DHID + j];
        const float hp  = hprev_f ? hprev_f[(size_t)b * DHID + j] : 0.0f;

        const float rg = sigmoid_f(gir + accr[r] + br_b);
        const float zg = sigmoid_f(giz + accz[r] + bz_b);
        const float ng = tanhf(gin + rg * (accn[r] + bn_b));
        const float hn = (1.0f - zg) * ng + zg * hp;

        out_t[(size_t)b * DHID + j] = hn;
        hnext_b[(size_t)b * DHID + j] = (bf16_t)hn;
        if (hlast) hlast[(size_t)b * DHID + j] = hn;
    }
}

// ---------------------------------------------------------------------------
extern "C" void kernel_launch(void* const* d_in, const int* in_sizes, int n_in,
                              void* d_out, int out_size, void* d_ws, size_t ws_size,
                              hipStream_t stream) {
    const float* x    = (const float*)d_in[0];   // (T, B, DIN)
    const float* W_ih = (const float*)d_in[1];   // (G3, DIN)
    const float* W_hh = (const float*)d_in[2];   // (G3, DHID)
    const float* b_ih = (const float*)d_in[3];   // (G3)
    const float* b_hh = (const float*)d_in[4];   // (G3)
    float* out = (float*)d_out;                  // (T*B*DHID) out + (B*DHID) h_last

    // ---- workspace carve-up ----
    char* ws = (char*)d_ws;
    size_t off = 0;
    bf16_t* xb   = (bf16_t*)(ws + off); off += (size_t)T_SEQ * BATCH * DIN * 2;
    bf16_t* wihb = (bf16_t*)(ws + off); off += (size_t)G3 * DIN * 2;
    bf16_t* whhb = (bf16_t*)(ws + off); off += (size_t)G3 * DHID * 2;
    bf16_t* hb0  = (bf16_t*)(ws + off); off += (size_t)BATCH * DHID * 2;
    bf16_t* hb1  = (bf16_t*)(ws + off); off += (size_t)BATCH * DHID * 2;
    off = (off + 255) & ~(size_t)255;
    float* gi    = (float*)(ws + off);  // (T*B, G3) fp32

    // ---- precision conversion ----
    {
        int n4 = (T_SEQ * BATCH * DIN) / 4;
        cvt_f32_to_bf16<<<(n4 + 255) / 256, 256, 0, stream>>>(x, xb, n4);
        n4 = (G3 * DIN) / 4;
        cvt_f32_to_bf16<<<(n4 + 255) / 256, 256, 0, stream>>>(W_ih, wihb, n4);
        n4 = (G3 * DHID) / 4;
        cvt_f32_to_bf16<<<(n4 + 255) / 256, 256, 0, stream>>>(W_hh, whhb, n4);
        const int nh = BATCH * DHID;
        zero_bf16<<<(nh + 255) / 256, 256, 0, stream>>>(hb0, nh);
    }

    // ---- gi = x @ W_ih^T + b_ih  (register-blocked WMMA GEMM) ----
    {
        const int blocks64 = (T_SEQ * BATCH / 64) * (G3 / 64);   // 24576 waves
        gi_gemm_kernel<<<blocks64 / 8, 256, 0, stream>>>(xb, wihb, b_ih, gi);
    }

    // ---- sequential GRU scan (512 small WMMA kernels, graph-captured) ----
    const size_t stepHf = (size_t)BATCH * DHID;
    const size_t stepGi = (size_t)BATCH * G3;
    for (int t = 0; t < T_SEQ; ++t) {
        bf16_t* hin  = (t & 1) ? hb1 : hb0;
        bf16_t* hout = (t & 1) ? hb0 : hb1;
        const float* hpf = (t == 0) ? nullptr : (out + (size_t)(t - 1) * stepHf);
        float* hlast = (t == T_SEQ - 1) ? (out + (size_t)T_SEQ * stepHf) : nullptr;
        gru_step_kernel<<<32, 256, 0, stream>>>(
            hin, hpf, whhb, b_hh, gi + (size_t)t * stepGi,
            out + (size_t)t * stepHf, hout, hlast);
    }
}
